// SparseMoE_2250562863537
// MI455X (gfx1250) — matrix operand order repaired and checked by reading, not compiled, not run on hardware
//
#include <hip/hip_runtime.h>
#include <stdint.h>

// ---------------------------------------------------------------------------
// Sparse MoE, top-1 of 8 experts.  B=8192 tokens, D=2048, 8 experts of
// [2048,2048].  out[b] = We[argmax_e gate] @ concat(x,y)[b] + be[e].
// Strategy: bf16 WMMA grouped-GEMM (tokens bucketed by expert), f32 accum.
// GEMM: 128x128 tile, K=32 per stage, double-buffered LDS -> ONE barrier per
// 8-WMMA stage, software-pipelined global->LDS staging (16 staged VGPRs,
// chosen to stay below the spill threshold observed at K=64).
// ws layout (assumes ws_size >= ~97 MB):
//   [0,32)        int cnt[8]
//   [256,256K)    int tokens[8][8192]
//   [512K, +32MB) bf16 inp[8192][2048]        (concat(x,y) converted)
//   [.., +64MB)   bf16 We[8][2048][2048]
// ---------------------------------------------------------------------------

typedef __attribute__((ext_vector_type(16))) __bf16 v16bf;
typedef __attribute__((ext_vector_type(8)))  float  v8f;

#define B_TOK 8192
#define L_DIM 1024
#define E_DIM 1024
#define D_DIM 2048
#define NEXP  8

#define BM 128
#define BN 128
#define BK 32
#define GEMM_THREADS 256
#define MAX_MTILES 72             // >= 8192/128 + 7 (worst-case padding tiles)

#define WS_CNT_OFF 0
#define WS_TOK_OFF 256
#define WS_INP_OFF ((size_t)(1u << 19))                         // 512 KB
#define WS_WE_OFF  (WS_INP_OFF + (size_t)B_TOK * D_DIM * 2)     // +32 MB

__device__ __forceinline__ unsigned short f2bf(float f) {
    unsigned int u = __float_as_uint(f);
    unsigned int r = u + 0x7FFFu + ((u >> 16) & 1u);   // round-to-nearest-even
    return (unsigned short)(r >> 16);
}

union FragBF { uint4 q[2]; v16bf v; };

// ---------------------------------------------------------------- kernel 1 --
__global__ void zero_cnt_kernel(int* __restrict__ cnt) {
    if (threadIdx.x < NEXP) cnt[threadIdx.x] = 0;
}

// ---------------------------------------------------------------- kernel 2 --
// fp32 -> bf16 weight conversion: 8*2048*2048 = 33,554,432 elems, 8/thread.
__global__ __launch_bounds__(256) void convert_we_kernel(
    const float* __restrict__ We, unsigned short* __restrict__ web) {
    size_t i = ((size_t)blockIdx.x * 256 + threadIdx.x) * 8;
    float4 a = *(const float4*)(We + i);
    float4 b = *(const float4*)(We + i + 4);
    uint4 o;
    o.x = (unsigned)f2bf(a.x) | ((unsigned)f2bf(a.y) << 16);
    o.y = (unsigned)f2bf(a.z) | ((unsigned)f2bf(a.w) << 16);
    o.z = (unsigned)f2bf(b.x) | ((unsigned)f2bf(b.y) << 16);
    o.w = (unsigned)f2bf(b.z) | ((unsigned)f2bf(b.w) << 16);
    *(uint4*)(web + i) = o;
}

// ---------------------------------------------------------------- kernel 3 --
// Gate: one thread per token.  Computes 8 gate logits, argmax (lowest index
// wins ties, matching jax top_k), appends token to its expert bucket, and
// emits the bf16 concat(x,y) row.
__global__ __launch_bounds__(256) void gate_kernel(
    const float* __restrict__ x, const float* __restrict__ y,
    const float* __restrict__ gW, const float* __restrict__ gb,
    unsigned short* __restrict__ inp_bf,
    int* __restrict__ cnt, int* __restrict__ toks) {
    int b = blockIdx.x * 256 + threadIdx.x;
    if (b >= B_TOK) return;

    float acc[NEXP];
#pragma unroll
    for (int e = 0; e < NEXP; ++e) acc[e] = gb[e];

    const float* xr = x + (size_t)b * L_DIM;
    const float* yr = y + (size_t)b * E_DIM;
    unsigned short* orow = inp_bf + (size_t)b * D_DIM;

    for (int d = 0; d < L_DIM; d += 4) {
        float4 v = *(const float4*)(xr + d);
        orow[d + 0] = f2bf(v.x); orow[d + 1] = f2bf(v.y);
        orow[d + 2] = f2bf(v.z); orow[d + 3] = f2bf(v.w);
#pragma unroll
        for (int e = 0; e < NEXP; ++e) {
            const float* g = gW + (size_t)e * D_DIM + d;
            acc[e] += v.x * g[0] + v.y * g[1] + v.z * g[2] + v.w * g[3];
        }
    }
    for (int d = 0; d < E_DIM; d += 4) {
        float4 v = *(const float4*)(yr + d);
        orow[L_DIM + d + 0] = f2bf(v.x); orow[L_DIM + d + 1] = f2bf(v.y);
        orow[L_DIM + d + 2] = f2bf(v.z); orow[L_DIM + d + 3] = f2bf(v.w);
#pragma unroll
        for (int e = 0; e < NEXP; ++e) {
            const float* g = gW + (size_t)e * D_DIM + L_DIM + d;
            acc[e] += v.x * g[0] + v.y * g[1] + v.z * g[2] + v.w * g[3];
        }
    }

    int best = 0; float bv = acc[0];
#pragma unroll
    for (int e = 1; e < NEXP; ++e)
        if (acc[e] > bv) { bv = acc[e]; best = e; }

    int pos = atomicAdd(&cnt[best], 1);
    toks[best * B_TOK + pos] = b;
}

// ---------------------------------------------------------------- kernel 4 --
// Grouped GEMM: C[tok, o] = sum_k A[tok,k] * We[e][o,k] + be[e][o]
__global__ __launch_bounds__(GEMM_THREADS) void moe_gemm_kernel(
    const unsigned short* __restrict__ inp_bf,
    const unsigned short* __restrict__ web,
    const float* __restrict__ be,
    const int* __restrict__ cnt,
    const int* __restrict__ toks,
    float* __restrict__ out) {
    // Double-buffered staging: 2 x (128x32) bf16 for A and B = 32 KB total.
    __shared__ __attribute__((aligned(16))) unsigned short sA[2][BM * BK];
    __shared__ __attribute__((aligned(16))) unsigned short sB[2][BN * BK];
    __shared__ int rowid[BM];

    // Resolve (expert, local m-tile) from dynamic per-expert counts.
    int mt = blockIdx.x;
    int e = -1, lm = 0, ce = 0;
#pragma unroll
    for (int i = 0; i < NEXP; ++i) {
        int ci = cnt[i];
        int t = (ci + BM - 1) / BM;
        if (e < 0) {
            if (mt < t) { e = i; lm = mt; ce = ci; }
            else mt -= t;
        }
    }
    if (e < 0) return;   // uniform across the block

    const int n0  = blockIdx.y * BN;
    const int tid = threadIdx.x;
    const int lane = tid & 31;
    const int w = tid >> 5;

    if (tid < BM) {
        int idx = lm * BM + tid;
        rowid[tid] = (idx < ce) ? toks[e * B_TOK + idx] : -1;
    }
    __syncthreads();

    v8f vzero = {0.f, 0.f, 0.f, 0.f, 0.f, 0.f, 0.f, 0.f};
    v8f acc[8];
#pragma unroll
    for (int i = 0; i < 8; ++i) acc[i] = vzero;

    // Global->LDS staging: 256 threads, 16 bf16 (32B) per matrix each.
    const int lr = tid >> 1;            // row 0..127
    const int lc = (tid & 1) * 16;      // half-row offset 0 or 16
    const int mytok = rowid[lr];        // hoisted: one ds_load total
    const unsigned short* arowp =
        (mytok >= 0) ? (inp_bf + (size_t)mytok * D_DIM + lc) : (const unsigned short*)0;
    const unsigned short* qrow =
        web + ((size_t)e * D_DIM + n0 + lr) * D_DIM + lc;

    // Frag addressing (row/col = lane%16, k = 16*(lane/16) + s for both A and B).
    const int arow = w * 16 + (lane & 15);
    const int ka = (lane >> 4) * 16;    // A frag: K = ka..ka+15 contiguous
    const int kb = (lane >> 4) * 16;    // B frag: K = kb..kb+15 contiguous

    const uint4 z4 = make_uint4(0u, 0u, 0u, 0u);
    uint4 ra0 = z4, ra1 = z4, rb0, rb1;

    // Prologue: fetch k=0 and commit to buffer 0.
    if (arowp) { ra0 = *(const uint4*)(arowp); ra1 = *(const uint4*)(arowp + 8); }
    rb0 = *(const uint4*)(qrow);
    rb1 = *(const uint4*)(qrow + 8);
    *(uint4*)(&sA[0][lr * BK + lc])     = ra0;
    *(uint4*)(&sA[0][lr * BK + lc + 8]) = ra1;
    *(uint4*)(&sB[0][lr * BK + lc])     = rb0;
    *(uint4*)(&sB[0][lr * BK + lc + 8]) = rb1;
    __syncthreads();

#pragma unroll 1
    for (int k0 = 0; k0 < D_DIM; k0 += BK) {
        const int buf = (k0 / BK) & 1;
        const int kn = k0 + BK;

        // ---- issue global fetch for next stage (hidden under WMMAs) ----
        if (kn < D_DIM) {
            ra0 = z4; ra1 = z4;
            if (arowp) {
                ra0 = *(const uint4*)(arowp + kn);
                ra1 = *(const uint4*)(arowp + kn + 8);
            }
            rb0 = *(const uint4*)(qrow + kn);
            rb1 = *(const uint4*)(qrow + kn + 8);
        }

        // ---- compute: 8 WMMAs from current buffer ----
        FragBF fa;
        fa.q[0] = *(const uint4*)(&sA[buf][arow * BK + ka]);
        fa.q[1] = *(const uint4*)(&sA[buf][arow * BK + ka + 8]);
#pragma unroll
        for (int nt = 0; nt < 8; ++nt) {
            FragBF fb;
            const int brow = nt * 16 + (lane & 15);
            fb.q[0] = *(const uint4*)(&sB[buf][brow * BK + kb]);
            fb.q[1] = *(const uint4*)(&sB[buf][brow * BK + kb + 8]);
            acc[nt] = __builtin_amdgcn_wmma_f32_16x16x32_bf16(
                false, fa.v, false, fb.v, (short)0, acc[nt], false, false);
        }

        // ---- commit next stage to the other buffer; single barrier ----
        if (kn < D_DIM) {
            *(uint4*)(&sA[buf ^ 1][lr * BK + lc])     = ra0;
            *(uint4*)(&sA[buf ^ 1][lr * BK + lc + 8]) = ra1;
            *(uint4*)(&sB[buf ^ 1][lr * BK + lc])     = rb0;
            *(uint4*)(&sB[buf ^ 1][lr * BK + lc + 8]) = rb1;
        }
        __syncthreads();
    }

    // ---- epilogue: bias + scatter to token rows ----
#pragma unroll
    for (int nt = 0; nt < 8; ++nt) {
        const int o = n0 + nt * 16 + (lane & 15);
        const float bias = be[(size_t)e * D_DIM + o];
#pragma unroll
        for (int j = 0; j < 8; ++j) {
            const int m = j + ((lane >> 4) << 3);   // C layout: VGPR j -> M=j / j+8
            const int tok = rowid[w * 16 + m];
            if (tok >= 0)
                out[(size_t)tok * D_DIM + o] = acc[nt][j] + bias;
        }
    }
}

// ---------------------------------------------------------------------------
extern "C" void kernel_launch(void* const* d_in, const int* in_sizes, int n_in,
                              void* d_out, int out_size, void* d_ws, size_t ws_size,
                              hipStream_t stream) {
    (void)in_sizes; (void)n_in; (void)out_size; (void)ws_size;
    const float* x  = (const float*)d_in[0];
    const float* y  = (const float*)d_in[1];
    const float* We = (const float*)d_in[2];
    const float* be = (const float*)d_in[3];
    const float* gW = (const float*)d_in[4];
    const float* gb = (const float*)d_in[5];
    float* out = (float*)d_out;

    char* ws = (char*)d_ws;
    int* cnt  = (int*)(ws + WS_CNT_OFF);
    int* toks = (int*)(ws + WS_TOK_OFF);
    unsigned short* inp_bf = (unsigned short*)(ws + WS_INP_OFF);
    unsigned short* web    = (unsigned short*)(ws + WS_WE_OFF);

    zero_cnt_kernel<<<1, 32, 0, stream>>>(cnt);
    convert_we_kernel<<<(NEXP * D_DIM * D_DIM) / (256 * 8), 256, 0, stream>>>(We, web);
    gate_kernel<<<B_TOK / 256, 256, 0, stream>>>(x, y, gW, gb, inp_bf, cnt, toks);
    moe_gemm_kernel<<<dim3(MAX_MTILES, D_DIM / BN), GEMM_THREADS, 0, stream>>>(
        inp_bf, web, be, cnt, toks, out);
}
